// DTAM_89275190214682
// MI455X (gfx1250) — matrix operand order repaired and checked by reading, not compiled, not run on hardware
//
#include <hip/hip_runtime.h>
#include <stdint.h>

typedef unsigned short u16;
typedef unsigned int   u32;

typedef __attribute__((ext_vector_type(16))) __bf16 v16bf;
typedef __attribute__((ext_vector_type(8)))  float  v8f;
typedef __attribute__((ext_vector_type(8)))  u32    v8u;

#define DEV static __device__ __forceinline__

// ---------------- constants ----------------
static constexpr int    Bn   = 4;
static constexpr int    Cc   = 192;     // C
static constexpr int    C2   = 384;     // 2C
static constexpr int    HC   = 96;      // head channels
static constexpr int    HW   = 16384;   // pixels per image
static constexpr int    PT   = 65536;   // total pixels (B*HW)
static constexpr size_t TS   = (size_t)C2 * PT;  // elements per q/k/v slab

// workspace layout (bytes, all 16B aligned)
static constexpr size_t OFF_XN   = 0;                          // 65536*192*2
static constexpr size_t OFF_WBF  = 25165824;                   // 1152*192*2
static constexpr size_t OFF_WOBF = 25608192;                   // 192*384*2
static constexpr size_t OFF_LAM  = 25755648;                   // scalar
static constexpr size_t OFF_G    = 25755904;                   // 16*96*96*4
static constexpr size_t OFF_ATTN = 26345728;                   // 8*96*96*2
static constexpr size_t OFF_PRE  = 26493184;                   // 3*384*65536*2
static constexpr size_t OFF_POST = 177488128;                  // 3*384*65536*2
static constexpr size_t OFF_Y    = OFF_PRE;                    // reuse (fp32, 384*65536*4)
static constexpr size_t OFF_YNT  = OFF_PRE + 100663296;        // bf16 65536*384

// ---------------- helpers ----------------
DEV float bf2f(u16 u) { return __uint_as_float(((u32)u) << 16); }
DEV u16   f2bf(float f) {
  u32 u = __float_as_uint(f);
  u32 r = u + 0x7FFFu + ((u >> 16) & 1u);   // round to nearest even
  return (u16)(r >> 16);
}
DEV v8f zero8() { v8f z; for (int i = 0; i < 8; ++i) z[i] = 0.f; return z; }

DEV uint4 pack_bf8(v8f a) {
  uint4 r;
  r.x = (u32)f2bf(a[0]) | ((u32)f2bf(a[1]) << 16);
  r.y = (u32)f2bf(a[2]) | ((u32)f2bf(a[3]) << 16);
  r.z = (u32)f2bf(a[4]) | ((u32)f2bf(a[5]) << 16);
  r.w = (u32)f2bf(a[6]) | ((u32)f2bf(a[7]) << 16);
  return r;
}

// A-matrix fragment, 16x32 bf16 (ISA 7.12.2):
//   lanes 0-15 : M=lane,    elems = K{k0..k0+7, k0+16..k0+23}
//   lanes16-31 : M=lane-16, elems = K{k0+8..15, k0+24..31}
DEV v16bf ld_frag_a(const u16* A, int ldk, int m0, int k0, int lane) {
  int m = lane & 15, hi = lane >> 4;
  const u16* r = A + (size_t)(m0 + m) * ldk + k0 + 8 * hi;
  uint4 c0 = *(const uint4*)r;
  uint4 c1 = *(const uint4*)(r + 16);
  v8u u; u[0]=c0.x; u[1]=c0.y; u[2]=c0.z; u[3]=c0.w; u[4]=c1.x; u[5]=c1.y; u[6]=c1.z; u[7]=c1.w;
  return __builtin_bit_cast(v16bf, u);
}

// B-matrix fragment, 32x16 bf16 (column n contiguous in K in our storage):
//   lanes 0-15 : N=lane,    elems = K{k0..k0+7, k0+16..k0+23}
//   lanes16-31 : N=lane-16, elems = K{k0+8..15, k0+24..31}
// Bsrc is "row per output column" storage with K-stride ldk.
DEV v16bf ld_frag_b(const u16* Bsrc, int ldk, int n0, int k0, int lane) {
  int n = lane & 15, hi = lane >> 4;
  const u16* r = Bsrc + (size_t)(n0 + n) * ldk + k0 + 8 * hi;
  uint4 c0 = *(const uint4*)r;
  uint4 c1 = *(const uint4*)(r + 16);
  v8u u; u[0]=c0.x; u[1]=c0.y; u[2]=c0.z; u[3]=c0.w; u[4]=c1.x; u[5]=c1.y; u[6]=c1.z; u[7]=c1.w;
  return __builtin_bit_cast(v16bf, u);
}

DEV v8f wmma_bf16(v16bf a, v16bf b, v8f c) {
  return __builtin_amdgcn_wmma_f32_16x16x32_bf16(false, a, false, b, (short)0, c, false, false);
}

// ---------------- small prep kernels ----------------
__global__ void k_convert_w(const float* __restrict__ Wq, const float* __restrict__ Wk,
                            const float* __restrict__ Wv, const float* __restrict__ Wo,
                            u16* __restrict__ wbf, u16* __restrict__ wobf) {
  int i = blockIdx.x * 256 + threadIdx.x;
  if (i < 1152 * 192) {
    int row = i / 192, c = i - row * 192;
    const float* src = (row < 384) ? Wq : ((row < 768) ? Wk : Wv);
    int o = row % 384;
    wbf[i] = f2bf(src[o * 192 + c]);
  } else {
    int j = i - 1152 * 192;
    if (j < 192 * 384) wobf[j] = f2bf(Wo[j]);
  }
}

__global__ void k_lam(const float* lq1, const float* lk1,
                      const float* lq2, const float* lk2, float* lam) {
  __shared__ float s1[128], s2[128];
  int t = threadIdx.x;
  float a = 0.f, b = 0.f;
  if (t < HC) { a = lq1[t] * lk1[t]; b = lq2[t] * lk2[t]; }
  s1[t] = a; s2[t] = b; __syncthreads();
  for (int o = 64; o > 0; o >>= 1) {
    if (t < o) { s1[t] += s1[t + o]; s2[t] += s2[t + o]; }
    __syncthreads();
  }
  if (t == 0) *lam = __expf(s1[0]) - __expf(s2[0]) + 0.8f;
}

__global__ void k_zero(float* p, int n) {
  int i = blockIdx.x * 256 + threadIdx.x;
  if (i < n) p[i] = 0.f;
}

// ---------------- stage 1: channel var-norm, transpose to pixel-major bf16 ----------------
__global__ void k_norm1(const float* __restrict__ x, const float* __restrict__ nw,
                        u16* __restrict__ xn) {
  __shared__ float ps1[256], ps2[256];
  __shared__ float scale[64];
  __shared__ __align__(16) u16 tile[64 * Cc];
  int t = threadIdx.x, px = t & 63, cy = t >> 6;
  int p0 = blockIdx.x * 64;            // 64 consecutive pixels, same image
  int b = p0 >> 14;
  int hw = (p0 & 16383) + px;
  const float* xb = x + (size_t)b * Cc * HW + hw;
  float s1 = 0.f, s2 = 0.f;
  for (int ch = cy; ch < Cc; ch += 4) {
    float v = xb[(size_t)ch * HW];
    s1 += v; s2 += v * v;
  }
  ps1[t] = s1; ps2[t] = s2; __syncthreads();
  if (t < 64) {
    float a = ps1[t] + ps1[t + 64] + ps1[t + 128] + ps1[t + 192];
    float q = ps2[t] + ps2[t + 64] + ps2[t + 128] + ps2[t + 192];
    float mean = a * (1.f / 192.f);
    float var  = q * (1.f / 192.f) - mean * mean;
    scale[t] = rsqrtf(var + 1e-5f);
  }
  __syncthreads();
  float sc = scale[px];
  for (int ch = cy; ch < Cc; ch += 4)
    tile[px * Cc + ch] = f2bf(xb[(size_t)ch * HW] * sc * nw[ch]);
  __syncthreads();
  uint4* dst = (uint4*)(xn + (size_t)p0 * Cc);          // contiguous 64*192 bf16
  const uint4* src = (const uint4*)tile;
  for (int i = t; i < 64 * Cc / 8; i += 256) dst[i] = src[i];
}

// ---------------- stage 2: fused QKV GEMM  (M=65536, N=1152, K=192) ----------------
__global__ void k_gemm_qkv(const u16* __restrict__ xn, const u16* __restrict__ wbf,
                           u16* __restrict__ pre) {
  int lane = threadIdx.x & 31, w = threadIdx.x >> 5;
  int m0 = blockIdx.x * 128 + w * 16;
  int n0 = blockIdx.y * 64;
  v8f acc[4]; for (int j = 0; j < 4; ++j) acc[j] = zero8();
  for (int k0 = 0; k0 < Cc; k0 += 32) {
    if (k0 + 32 < Cc)
      __builtin_prefetch((const char*)(xn + (size_t)(m0 + (lane & 15)) * Cc + k0 + 32), 0, 3);
    v16bf a = ld_frag_a(xn, Cc, m0, k0, lane);
#pragma unroll
    for (int j = 0; j < 4; ++j) {
      v16bf bf = ld_frag_b(wbf, Cc, n0 + j * 16, k0, lane);
      acc[j] = wmma_bf16(a, bf, acc[j]);
    }
  }
  int hi = lane >> 4;
  int pbase = m0 + 8 * hi;
#pragma unroll
  for (int j = 0; j < 4; ++j) {
    int nc = n0 + j * 16 + (lane & 15);
    int tsel = nc / C2, o = nc % C2;
    u16* dst = pre + (size_t)tsel * TS + (size_t)o * PT + pbase;  // channel-major
    *(uint4*)dst = pack_bf8(acc[j]);
  }
}

// ---------------- stage 3a: depthwise 3x3 on q,k (channel-major in & out) ----------------
__global__ void k_dwconv_qk(const u16* __restrict__ pre, const float* __restrict__ Dq,
                            const float* __restrict__ Dk, u16* __restrict__ post) {
  int t = threadIdx.x;
  int hw = blockIdx.x * 256 + t;
  int plane = blockIdx.y;                 // [0, 2*384*4)
  int tsel = plane / 1536;
  int rem = plane - tsel * 1536;
  int o = rem >> 2, b = rem & 3;
  const float* Wd = (tsel == 0 ? Dq : Dk) + o * 9;
  const u16* in = pre + (size_t)tsel * TS + (size_t)o * PT + b * HW;
  int xx = hw & 127, yy = hw >> 7;
  float s = 0.f;
#pragma unroll
  for (int ky = 0; ky < 3; ++ky) {
    int iy = yy + ky - 1;
    if (iy < 0 || iy > 127) continue;
#pragma unroll
    for (int kx = 0; kx < 3; ++kx) {
      int ix = xx + kx - 1;
      if (ix < 0 || ix > 127) continue;
      s += bf2f(in[iy * 128 + ix]) * Wd[ky * 3 + kx];
    }
  }
  post[(size_t)tsel * TS + (size_t)o * PT + b * HW + hw] = f2bf(s);
}

// ---------------- stage 3b: depthwise 3x3 on v, output transposed pixel-major ----------------
__global__ void k_dwconv_v(const u16* __restrict__ pre, const float* __restrict__ Dv,
                           u16* __restrict__ post) {
  __shared__ __align__(16) u16 tile[64 * Cc];
  int t = threadIdx.x, px = t & 63, cy = t >> 6;
  int b2h = blockIdx.y;                   // b*2+h
  int b = b2h >> 1, h = b2h & 1;
  int hw = blockIdx.x * 64 + px;
  int xx = hw & 127, yy = hw >> 7;
  for (int ch = cy; ch < Cc; ch += 4) {
    int o = h * Cc + ch;
    const float* Wd = Dv + o * 9;
    const u16* in = pre + 2 * TS + (size_t)o * PT + b * HW;
    float s = 0.f;
#pragma unroll
    for (int ky = 0; ky < 3; ++ky) {
      int iy = yy + ky - 1;
      if (iy < 0 || iy > 127) continue;
#pragma unroll
      for (int kx = 0; kx < 3; ++kx) {
        int ix = xx + kx - 1;
        if (ix < 0 || ix > 127) continue;
        s += bf2f(in[iy * 128 + ix]) * Wd[ky * 3 + kx];
      }
    }
    tile[px * Cc + ch] = f2bf(s);
  }
  __syncthreads();
  // v_t[b2h][pixel][192 channels] — contiguous copy
  uint4* dst = (uint4*)(post + 2 * TS + ((size_t)b2h * HW + blockIdx.x * 64) * Cc);
  const uint4* src = (const uint4*)tile;
  for (int i = t; i < 64 * Cc / 8; i += 256) dst[i] = src[i];
}

// ---------------- stage 4: Gram matrices with split-K (M=N=96, K=16384) ----------------
__global__ void k_gram(const u16* __restrict__ post, float* __restrict__ G) {
  int lane = threadIdx.x & 31, w = threadIdx.x >> 5;      // 6 waves
  int gx = blockIdx.x;                                    // b*4 + h*2 + sel
  int b = gx >> 2, h = (gx >> 1) & 1, sel = gx & 1;
  int kc = blockIdx.y;                                    // 16 K-chunks of 1024
  size_t chbase = (size_t)(h * Cc + sel * HC) * PT + (size_t)b * HW;
  const u16* Q = post + chbase;
  const u16* K = post + TS + chbase;
  int m0 = w * 16;
  v8f acc[6]; for (int j = 0; j < 6; ++j) acc[j] = zero8();
  for (int s = 0; s < 32; ++s) {
    int k0 = kc * 1024 + s * 32;
    v16bf a = ld_frag_a(Q, PT, m0, k0, lane);
#pragma unroll
    for (int j = 0; j < 6; ++j) {
      v16bf bf = ld_frag_b(K, PT, j * 16, k0, lane);
      acc[j] = wmma_bf16(a, bf, acc[j]);
    }
  }
  float* Gb = G + (size_t)gx * 9216;
  int hi = lane >> 4, col0 = lane & 15;
#pragma unroll
  for (int j = 0; j < 6; ++j)
#pragma unroll
    for (int r = 0; r < 8; ++r)
      atomicAdd(&Gb[(m0 + 8 * hi + r) * 96 + j * 16 + col0], acc[j][r]);
}

// ---------------- stage 5: softmax + differential combine -> bf16 attn ----------------
__global__ void k_softmax(const float* __restrict__ G, const float* __restrict__ t1,
                          const float* __restrict__ t2, const float* __restrict__ lamp,
                          u16* __restrict__ attn) {
  __shared__ float r1[128], r2[128];
  int blk = blockIdx.x;
  int row = blk % 96, bh = blk / 96, h = bh & 1;
  const float* g1 = G + ((size_t)bh * 2 + 0) * 9216 + row * 96;
  const float* g2 = G + ((size_t)bh * 2 + 1) * 9216 + row * 96;
  int t = threadIdx.x;
  float v1 = -1e30f, v2 = -1e30f;
  if (t < 96) { v1 = g1[t] * t1[h]; v2 = g2[t] * t2[h]; }
  r1[t] = v1; r2[t] = v2; __syncthreads();
  for (int o = 64; o > 0; o >>= 1) {
    if (t < o) { r1[t] = fmaxf(r1[t], r1[t + o]); r2[t] = fmaxf(r2[t], r2[t + o]); }
    __syncthreads();
  }
  float m1 = r1[0], m2 = r2[0];
  __syncthreads();
  float e1 = (t < 96) ? __expf(v1 - m1) : 0.f;
  float e2 = (t < 96) ? __expf(v2 - m2) : 0.f;
  r1[t] = e1; r2[t] = e2; __syncthreads();
  for (int o = 64; o > 0; o >>= 1) {
    if (t < o) { r1[t] += r1[t + o]; r2[t] += r2[t + o]; }
    __syncthreads();
  }
  if (t < 96) {
    float v = e1 / r1[0] - (*lamp) * (e2 / r2[0]);
    attn[(size_t)bh * 9216 + row * 96 + t] = f2bf(v);
  }
}

// ---------------- stage 6: y = attn @ v  (M=96, N=16384, K=96) ----------------
__global__ void k_ygemm(const u16* __restrict__ attn, const u16* __restrict__ post,
                        float* __restrict__ Y) {
  int lane = threadIdx.x & 31, w = threadIdx.x >> 5;      // 8 waves
  int gy = blockIdx.y;
  int b = gy >> 2, h = (gy >> 1) & 1, sel = gy & 1;
  int b2h = b * 2 + h;
  int n0 = blockIdx.x * 128 + w * 16;
  const u16* A  = attn + (size_t)b2h * 9216;              // 96x96, row stride 96
  const u16* Vt = post + 2 * TS + (size_t)b2h * HW * Cc;  // [pixel][192ch]
  v8f acc[6]; for (int j = 0; j < 6; ++j) acc[j] = zero8();
  for (int k0 = 0; k0 < HC; k0 += 32) {
    v16bf bf = ld_frag_b(Vt, Cc, n0, sel * HC + k0, lane);
#pragma unroll
    for (int mi = 0; mi < 6; ++mi) {
      v16bf a = ld_frag_a(A, HC, mi * 16, k0, lane);
      acc[mi] = wmma_bf16(a, bf, acc[mi]);
    }
  }
  float* Yb = Y + ((size_t)b2h * Cc + sel * HC) * HW;
  int hi = lane >> 4, col = n0 + (lane & 15);
#pragma unroll
  for (int mi = 0; mi < 6; ++mi)
#pragma unroll
    for (int r = 0; r < 8; ++r)
      Yb[(size_t)(mi * 16 + 8 * hi + r) * HW + col] = acc[mi][r];
}

// ---------------- stage 7: head RMS norm, scale, transpose to pixel-major bf16 ----------------
__global__ void k_norm2(const float* __restrict__ Y, const float* __restrict__ hnw,
                        u16* __restrict__ ynt) {
  __shared__ float ps[256];
  __shared__ float scale[64];
  __shared__ __align__(16) u16 tile[64 * Cc];
  int t = threadIdx.x, px = t & 63, cy = t >> 6;
  int b2h = blockIdx.y;
  int b = b2h >> 1, h = b2h & 1;
  int n0 = blockIdx.x * 64;
  const float* Yb = Y + (size_t)b2h * Cc * HW + n0 + px;
  float ss = 0.f;
  for (int ch = cy; ch < Cc; ch += 4) {
    float v = Yb[(size_t)ch * HW];
    ss += v * v;
  }
  ps[t] = ss; __syncthreads();
  if (t < 64) {
    float q = ps[t] + ps[t + 64] + ps[t + 128] + ps[t + 192];
    scale[t] = rsqrtf(q * (1.f / 192.f) + 1e-6f) * (1.0f - 0.8f);
  }
  __syncthreads();
  float sc = scale[px];
  for (int ch = cy; ch < Cc; ch += 4)
    tile[px * Cc + ch] = f2bf(Yb[(size_t)ch * HW] * sc * hnw[h * Cc + ch]);
  __syncthreads();
  // ynt[pixel][384]; this block fills the 192-wide slice for head h
  for (int i = t; i < 64 * 24; i += 256) {
    int row = i / 24, piece = i - row * 24;
    uint4* dst = (uint4*)(ynt + ((size_t)(b * HW + n0 + row)) * C2 + h * Cc);
    dst[piece] = ((const uint4*)(tile + row * Cc))[piece];
  }
}

// ---------------- stage 8: output GEMM + residual (M=65536, N=192, K=384) ----------------
__global__ void k_gemm_out(const u16* __restrict__ ynt, const u16* __restrict__ wobf,
                           const float* __restrict__ xres, float* __restrict__ out) {
  int lane = threadIdx.x & 31, w = threadIdx.x >> 5;
  int m0 = blockIdx.x * 128 + w * 16;
  int n0 = blockIdx.y * 64;
  v8f acc[4]; for (int j = 0; j < 4; ++j) acc[j] = zero8();
  for (int k0 = 0; k0 < C2; k0 += 32) {
    if (k0 + 32 < C2)
      __builtin_prefetch((const char*)(ynt + (size_t)(m0 + (lane & 15)) * C2 + k0 + 32), 0, 3);
    v16bf a = ld_frag_a(ynt, C2, m0, k0, lane);
#pragma unroll
    for (int j = 0; j < 4; ++j) {
      v16bf bf = ld_frag_b(wobf, C2, n0 + j * 16, k0, lane);
      acc[j] = wmma_bf16(a, bf, acc[j]);
    }
  }
  int hi = lane >> 4;
  int p = m0 + 8 * hi;
  int b = p >> 14, hwb = p & 16383;
#pragma unroll
  for (int j = 0; j < 4; ++j) {
    int o = n0 + j * 16 + (lane & 15);
    size_t base = ((size_t)b * Cc + o) * HW + hwb;
    float4 r0 = *(const float4*)(xres + base);
    float4 r1 = *(const float4*)(xres + base + 4);
    r0.x += acc[j][0]; r0.y += acc[j][1]; r0.z += acc[j][2]; r0.w += acc[j][3];
    r1.x += acc[j][4]; r1.y += acc[j][5]; r1.z += acc[j][6]; r1.w += acc[j][7];
    *(float4*)(out + base) = r0;
    *(float4*)(out + base + 4) = r1;
  }
}

// ---------------- launch ----------------
extern "C" void kernel_launch(void* const* d_in, const int* in_sizes, int n_in,
                              void* d_out, int out_size, void* d_ws, size_t ws_size,
                              hipStream_t stream) {
  (void)in_sizes; (void)n_in; (void)out_size; (void)ws_size;
  const float* x   = (const float*)d_in[0];
  const float* nw  = (const float*)d_in[1];
  const float* Wq  = (const float*)d_in[2];
  const float* Wk  = (const float*)d_in[3];
  const float* Wv  = (const float*)d_in[4];
  const float* Dq  = (const float*)d_in[5];
  const float* Dk  = (const float*)d_in[6];
  const float* Dv  = (const float*)d_in[7];
  const float* t1  = (const float*)d_in[8];
  const float* t2  = (const float*)d_in[9];
  const float* hnw = (const float*)d_in[10];
  const float* Wo  = (const float*)d_in[11];
  const float* lq1 = (const float*)d_in[12];
  const float* lk1 = (const float*)d_in[13];
  const float* lq2 = (const float*)d_in[14];
  const float* lk2 = (const float*)d_in[15];
  float* out = (float*)d_out;
  char*  ws  = (char*)d_ws;

  u16*   XN   = (u16*)(ws + OFF_XN);
  u16*   WBF  = (u16*)(ws + OFF_WBF);
  u16*   WOBF = (u16*)(ws + OFF_WOBF);
  float* LAM  = (float*)(ws + OFF_LAM);
  float* G    = (float*)(ws + OFF_G);
  u16*   ATTN = (u16*)(ws + OFF_ATTN);
  u16*   PRE  = (u16*)(ws + OFF_PRE);
  u16*   POST = (u16*)(ws + OFF_POST);
  float* Y    = (float*)(ws + OFF_Y);     // reuses PRE region (dead after dwconv)
  u16*   YNT  = (u16*)(ws + OFF_YNT);

  k_convert_w<<<dim3(1152), dim3(256), 0, stream>>>(Wq, Wk, Wv, Wo, WBF, WOBF);
  k_lam<<<dim3(1), dim3(128), 0, stream>>>(lq1, lk1, lq2, lk2, LAM);
  k_norm1<<<dim3(1024), dim3(256), 0, stream>>>(x, nw, XN);
  k_gemm_qkv<<<dim3(512, 18), dim3(256), 0, stream>>>(XN, WBF, PRE);
  k_dwconv_qk<<<dim3(64, 3072), dim3(256), 0, stream>>>(PRE, Dq, Dk, POST);
  k_dwconv_v<<<dim3(256, 8), dim3(256), 0, stream>>>(PRE, Dv, POST);
  k_zero<<<dim3(576), dim3(256), 0, stream>>>(G, 16 * 96 * 96);
  k_gram<<<dim3(16, 16), dim3(192), 0, stream>>>(POST, G);
  k_softmax<<<dim3(768), dim3(128), 0, stream>>>(G, t1, t2, LAM, ATTN);
  k_ygemm<<<dim3(128, 16), dim3(256), 0, stream>>>(ATTN, POST, Y);
  k_norm2<<<dim3(256, 8), dim3(256), 0, stream>>>(Y, hnw, YNT);
  k_gemm_out<<<dim3(512, 3), dim3(256), 0, stream>>>(YNT, WOBF, x, out);
}
